// BiLSTM_Encoder_37022618091822
// MI455X (gfx1250) — matrix operand order repaired and checked by reading, not compiled, not run on hardware
//
#include <hip/hip_runtime.h>
#include <hip/hip_bf16.h>
#include <stddef.h>

// ---------------------------------------------------------------------------
// BiLSTM encoder for MI455X (gfx1250, wave32, WMMA).
// Phase 1: bf16 WMMA input projection, 32x32 register tile per wave,
//          A staged through LDS with async global->LDS loads (ASYNCcnt).
// Phase 2: 256 sequential step kernels, 16x16x1024 WMMA GEMMs per
//          (dir,gate,batch-half) against L2-resident W_hh.
// ---------------------------------------------------------------------------

typedef __attribute__((ext_vector_type(8)))  __bf16 v8bf;
typedef __attribute__((ext_vector_type(16))) __bf16 v16bf;
typedef __attribute__((ext_vector_type(8)))  float  v8f;

#define NB   32      // batch
#define SQ   256     // sequence length
#define DK   1024    // input dim
#define HD   1024    // hidden dim

__device__ __forceinline__ unsigned short f32_to_bf16(float f) {
    unsigned int u = __float_as_uint(f);
    unsigned int r = u + 0x7FFFu + ((u >> 16) & 1u);   // round-to-nearest-even
    return (unsigned short)(r >> 16);
}

__device__ __forceinline__ v16bf make_a_frag(const unsigned short* base) {
    // base -> this lane's first 8-element K chunk; second chunk at +16 elems.
    v8bf lo = *(const v8bf*)(base);
    v8bf hi = *(const v8bf*)(base + 16);
    return __builtin_shufflevector(lo, hi, 0, 1, 2, 3, 4, 5, 6, 7,
                                           8, 9, 10, 11, 12, 13, 14, 15);
}

// one async 16-byte global->LDS copy per lane (ASYNCcnt-tracked)
__device__ __forceinline__ void async_load16(unsigned lds_addr, unsigned long long ga) {
    asm volatile("global_load_async_to_lds_b128 %0, %1, off"
                 :: "v"(lds_addr), "v"(ga) : "memory");
}

__device__ __forceinline__ void wait_async0() {
    asm volatile("s_wait_asynccnt 0x0" ::: "memory");
}

// ---------------------------------------------------------------------------
// f32 -> bf16 casts of x, W_ih, W_hh
// ---------------------------------------------------------------------------
__global__ void k_cast(const float* __restrict__ x,
                       const float* __restrict__ wih,
                       const float* __restrict__ whh,
                       unsigned short* __restrict__ xbf,
                       unsigned short* __restrict__ wihbf,
                       unsigned short* __restrict__ whhbf) {
    const size_t total = (size_t)8 * 1024 * 1024;
    for (size_t i = (size_t)blockIdx.x * blockDim.x + threadIdx.x;
         i < total; i += (size_t)gridDim.x * blockDim.x) {
        xbf[i]   = f32_to_bf16(x[i]);
        wihbf[i] = f32_to_bf16(wih[i]);
        whhbf[i] = f32_to_bf16(whh[i]);
    }
}

// ---------------------------------------------------------------------------
// broadcast h0/c0 over batch; seed ping buffer of bf16 h
// ---------------------------------------------------------------------------
__global__ void k_init(const float* __restrict__ h0f, const float* __restrict__ h0b,
                       const float* __restrict__ c0f, const float* __restrict__ c0b,
                       float* __restrict__ hcur, float* __restrict__ ccur,
                       unsigned short* __restrict__ hbf0) {
    int idx = blockIdx.x * blockDim.x + threadIdx.x;    // 0..65535
    if (idx >= 2 * NB * HD) return;
    int dir = idx >> 15;
    int j   = idx & (HD - 1);
    float h = dir ? h0b[j] : h0f[j];
    float c = dir ? c0b[j] : c0f[j];
    hcur[idx] = h;
    ccur[idx] = c;
    hbf0[idx] = f32_to_bf16(h);
}

// ---------------------------------------------------------------------------
// Input projection: xh[g][s][b][j] = sum_d x[b][s][d]*Wih[g][d][j] + bias[g][j]
// grid: (HD/128, 8192/64, 8 gates), block 256 (8 waves, 2x4 wave grid).
// Each wave: 32x32 output tile = 2x2 WMMA tiles, K-loop of 32.
// A tile (64 rows x 32 k, 4KB) double-buffered in LDS via async loads.
// ---------------------------------------------------------------------------
__global__ void k_proj(const unsigned short* __restrict__ xbf,   // [8192][DK]
                       const unsigned short* __restrict__ wih,   // [8][DK][HD]
                       const float* __restrict__ bias,           // [8][HD]
                       float* __restrict__ xh) {                 // [8][SQ][NB][HD]
    __shared__ unsigned short atile[2][64 * 32];                 // 2 x 4KB

    const int tid  = threadIdx.x;
    const int lane = tid & 31;
    const int w    = tid >> 5;
    const int wr   = w >> 2;                 // wave row (M)
    const int wc   = w & 3;                  // wave col (N)
    const int g    = blockIdx.z;
    const int mblk = blockIdx.y * 64;
    const int m0   = mblk + wr * 32;
    const int j0   = blockIdx.x * 128 + wc * 32;

    const unsigned short* B = wih + (size_t)g * DK * HD;

    // async-copy mapping: thread -> one 16B chunk of the 64x32 A tile
    const int arow = tid >> 2;               // 0..63
    const int achk = (tid & 3) * 8;          // element offset within row
    const unsigned long long gaBase =
        (unsigned long long)(const void*)(xbf + ((size_t)(mblk + arow) * DK + achk));
    const unsigned ldsA0 =
        (unsigned)(unsigned long long)(const void*)&atile[0][arow * 32 + achk];
    const unsigned ldsA1 =
        (unsigned)(unsigned long long)(const void*)&atile[1][arow * 32 + achk];

    async_load16(ldsA0, gaBase);             // prologue fill of buffer 0

    const v8f vz = {0.f, 0.f, 0.f, 0.f, 0.f, 0.f, 0.f, 0.f};
    v8f acc00 = vz, acc01 = vz, acc10 = vz, acc11 = vz;

    const int koff = (lane < 16) ? 0 : 8;    // A-frag K sub-chunk select
    const int fr0  = wr * 32 + (lane & 15);  // LDS row for M-subtile 0
    const int fr1  = fr0 + 16;               // LDS row for M-subtile 1

    for (int i = 0; i < DK / 32; ++i) {
        const int k0  = i * 32;
        const int cur = i & 1;

        wait_async0();                       // my fill of buf[cur] arrived
        __syncthreads();                     // everyone's fill done, prev reads done
        if (i + 1 < DK / 32)                 // overlap next fill with compute
            async_load16(cur ? ldsA0 : ldsA1,
                         gaBase + (unsigned long long)(k0 + 32) * 2);

        const unsigned short* at = atile[cur];
        v16bf a0 = make_a_frag(at + fr0 * 32 + koff);
        v16bf a1 = make_a_frag(at + fr1 * 32 + koff);

        const unsigned short* brow = B + (size_t)(k0 + lane) * HD + j0;
        const unsigned short* bcol = B + (size_t)k0 * HD + j0 + (lane & 15);
        v16bf b0, b1;
#pragma unroll
        for (int s = 0; s < 16; ++s) {
            const size_t kk = (size_t)(koff + (s & 7) + ((s >> 3) << 4)) * HD;
            b0[s] = __builtin_bit_cast(__bf16, bcol[kk]);
            b1[s] = __builtin_bit_cast(__bf16, bcol[kk + 16]);
        }
        if (k0 + 32 < DK)
            __builtin_prefetch(brow + (size_t)32 * HD, 0, 1);

        acc00 = __builtin_amdgcn_wmma_f32_16x16x32_bf16(false, a0, false, b0, (short)0, acc00, false, false);
        acc01 = __builtin_amdgcn_wmma_f32_16x16x32_bf16(false, a0, false, b1, (short)0, acc01, false, false);
        acc10 = __builtin_amdgcn_wmma_f32_16x16x32_bf16(false, a1, false, b0, (short)0, acc10, false, false);
        acc11 = __builtin_amdgcn_wmma_f32_16x16x32_bf16(false, a1, false, b1, (short)0, acc11, false, false);
    }

    // store: C layout -> VGPR r = row r (+8 for upper lanes), lane%16 = col
    const int rbase = (lane < 16) ? 0 : 8;
    const int colc  = lane & 15;
#pragma unroll
    for (int ms = 0; ms < 2; ++ms) {
        const v8f* cl = ms ? &acc10 : &acc00;
        const v8f* ch = ms ? &acc11 : &acc01;
#pragma unroll
        for (int r = 0; r < 8; ++r) {
            int m    = m0 + ms * 16 + r + rbase;
            int bidx = m >> 8;               // batch (m = b*256 + s)
            int sidx = m & 255;              // seq
            size_t rowbase = (((size_t)g * SQ + sidx) * NB + bidx) * HD;
            int c0i = j0 + colc;
            int c1i = j0 + 16 + colc;
            xh[rowbase + c0i] = (*cl)[r] + bias[g * HD + c0i];
            xh[rowbase + c1i] = (*ch)[r] + bias[g * HD + c1i];
        }
    }
}

// ---------------------------------------------------------------------------
// One LSTM time step, both directions.
// grid: (HD/16 = 64, 2 dirs), block 256 (8 waves).
// wave w: gate g = w&3, batch half mb = w>>2 -> 16x16x1024 GEMM on h_prev.
// ---------------------------------------------------------------------------
__global__ void k_step(const float* __restrict__ xh,            // [8][SQ][NB][HD]
                       const unsigned short* __restrict__ whh,  // [8][HD][HD] bf16
                       const float* __restrict__ mask,          // [NB][SQ]
                       const unsigned short* __restrict__ hin,  // [2][NB][HD] bf16
                       unsigned short* __restrict__ hout,       // [2][NB][HD] bf16
                       float* __restrict__ hcur,                // [2][NB][HD]
                       float* __restrict__ ccur,                // [2][NB][HD]
                       float* __restrict__ out,                 // hidden + final
                       int step) {
    __shared__ float pre[8 * 256];   // 8 tiles of 16x16 f32

    const int lane = threadIdx.x & 31;
    const int w    = threadIdx.x >> 5;
    const int dir  = blockIdx.y;
    const int j0   = blockIdx.x << 4;
    const int s    = dir ? (SQ - 1 - step) : step;
    const int g    = w & 3;
    const int mb   = w >> 2;

    const unsigned short* A = hin + (size_t)dir * NB * HD;
    const unsigned short* B = whh + (size_t)(dir * 4 + g) * HD * HD;

    v8f c = {0.f, 0.f, 0.f, 0.f, 0.f, 0.f, 0.f, 0.f};

    const int arow = mb * 16 + (lane & 15);
    const int koff = (lane < 16) ? 0 : 8;
    const unsigned short* Abase = A + (size_t)arow * HD + koff;

    for (int k0 = 0; k0 < HD; k0 += 32) {
        v16bf a = make_a_frag(Abase + k0);
        v16bf b;
#pragma unroll
        for (int s = 0; s < 16; ++s)
            b[s] = __builtin_bit_cast(__bf16, B[(size_t)(k0 + koff + (s & 7) + ((s >> 3) << 4)) * HD + j0 + (lane & 15)]);
        if (k0 + 32 < HD)
            __builtin_prefetch(B + (size_t)(k0 + 32 + lane) * HD + j0, 0, 1);
        c = __builtin_amdgcn_wmma_f32_16x16x32_bf16(
                false, a, false, b, (short)0, c, false, false);
    }

    float* pw = &pre[w * 256];
    const int colLoc = lane & 15;
    const int rbase  = (lane < 16) ? 0 : 8;
#pragma unroll
    for (int r = 0; r < 8; ++r)
        pw[(r + rbase) * 16 + colLoc] = c[r];

    __syncthreads();

    for (int e = threadIdx.x; e < NB * 16; e += blockDim.x) {
        const int b   = e >> 4;
        const int jc  = e & 15;
        const int j   = j0 + jc;
        const int mbl = b >> 4;
        const int rl  = b & 15;

        float p[4];
#pragma unroll
        for (int gg = 0; gg < 4; ++gg) {
            float acc = pre[(gg * 2 + mbl) * 256 + rl * 16 + jc];
            acc += xh[(((size_t)(dir * 4 + gg) * SQ + s) * NB + b) * HD + j];
            p[gg] = acc;
        }

        const float g0 = 1.f / (1.f + __expf(-p[0]));
        const float g1 = 1.f / (1.f + __expf(-p[1]));
        const float g2 = 1.f / (1.f + __expf(-p[2]));
        const float gc = tanhf(p[3]);

        const size_t hci = (size_t)dir * (NB * HD) + (size_t)b * HD + j;
        const float cprev = ccur[hci];
        const float hprev = hcur[hci];

        const float cnew = g0 * cprev + g1 * gc;
        const float m    = mask[b * SQ + s];
        const float hnew = (m == 0.f) ? hprev : g2 * tanhf(cnew);

        ccur[hci] = cnew;
        hcur[hci] = hnew;
        hout[hci] = f32_to_bf16(hnew);

        out[((size_t)b * SQ + s) * (2 * HD) + dir * HD + j] = hnew * m;

        if (step == SQ - 1)
            out[(size_t)NB * SQ * 2 * HD + (size_t)b * (2 * HD) + dir * HD + j] = hnew;
    }
}

// ---------------------------------------------------------------------------
// launch
// ---------------------------------------------------------------------------
extern "C" void kernel_launch(void* const* d_in, const int* in_sizes, int n_in,
                              void* d_out, int out_size, void* d_ws, size_t ws_size,
                              hipStream_t stream) {
    const float* x    = (const float*)d_in[0];
    const float* mask = (const float*)d_in[1];
    const float* wih  = (const float*)d_in[2];
    const float* whh  = (const float*)d_in[3];
    const float* bias = (const float*)d_in[4];
    const float* h0f  = (const float*)d_in[5];
    const float* h0b  = (const float*)d_in[6];
    const float* c0f  = (const float*)d_in[7];
    const float* c0b  = (const float*)d_in[8];
    float* out = (float*)d_out;
    char*  ws  = (char*)d_ws;

    const size_t SZ_BF  = (size_t)8 * 1024 * 1024 * 2;     // 16 MiB
    const size_t SZ_XH  = (size_t)8 * SQ * NB * HD * 4;    // 256 MiB
    const size_t SZ_ST  = (size_t)2 * NB * HD * 4;         // 256 KiB
    const size_t SZ_HBF = (size_t)2 * NB * HD * 2;         // 128 KiB

    unsigned short* wihbf = (unsigned short*)(ws);
    unsigned short* whhbf = (unsigned short*)(ws + SZ_BF);
    unsigned short* xbf   = (unsigned short*)(ws + 2 * SZ_BF);
    float*          xhbuf = (float*)(ws + 3 * SZ_BF);
    float*          hcur  = (float*)(ws + 3 * SZ_BF + SZ_XH);
    float*          ccur  = (float*)(ws + 3 * SZ_BF + SZ_XH + SZ_ST);
    unsigned short* hbf0  = (unsigned short*)(ws + 3 * SZ_BF + SZ_XH + 2 * SZ_ST);
    unsigned short* hbf1  = (unsigned short*)(ws + 3 * SZ_BF + SZ_XH + 2 * SZ_ST + SZ_HBF);

    k_cast<<<2048, 256, 0, stream>>>(x, wih, whh, xbf, wihbf, whhbf);
    k_init<<<256, 256, 0, stream>>>(h0f, h0b, c0f, c0b, hcur, ccur, hbf0);

    // 8192x1024x1024 per gate, 8 gates; 64x128 block tile, 32x32 wave tile
    k_proj<<<dim3(HD / 128, (NB * SQ) / 64, 8), 256, 0, stream>>>(xbf, wihbf, bias, xhbuf);

    // sequential recurrence: 256 step launches (graph-captured by harness)
    for (int t = 0; t < SQ; ++t) {
        unsigned short* hin  = (t & 1) ? hbf1 : hbf0;
        unsigned short* hout = (t & 1) ? hbf0 : hbf1;
        k_step<<<dim3(HD / 16, 2), 256, 0, stream>>>(
            xhbuf, whhbf, mask, hin, hout, hcur, ccur, out, t);
    }
}
